// SDGCN_with_ln_res_scaled_69295002353945
// MI455X (gfx1250) — matrix operand order repaired and checked by reading, not compiled, not run on hardware
//
#include <hip/hip_runtime.h>
#include <math.h>

// Problem constants from the reference: B=4, T=16, N=1024, D=128.
#define BT 64
#define NN 1024
#define DD 128

typedef __attribute__((ext_vector_type(16))) __bf16 v16bf;
typedef __attribute__((ext_vector_type(8)))  __bf16 v8bf;
typedef __attribute__((ext_vector_type(8)))  float  v8f;

// LDS pitches (elements). Chosen for 16B alignment of every vector access and
// conflict-free strides (272B and 80B row pitches).
#define XR_P 136   // row-major X chunk  [32][136]
#define XT_P 40    // d-major  X chunk   [128][40]
#define PS_P 136   // per-wave scratch   [16][136]

#define WMMA_BF16(a, b, c) \
  __builtin_amdgcn_wmma_f32_16x16x32_bf16(false, (a), false, (b), (short)0, (c), false, false)

static __device__ __forceinline__ v16bf cat8(v8bf a, v8bf b) {
  return __builtin_shufflevector(a, b, 0,1,2,3,4,5,6,7,8,9,10,11,12,13,14,15);
}
// 16 contiguous bf16 from LDS (two ds_read_b128)
static __device__ __forceinline__ v16bf lds16(const __bf16* p) {
  return cat8(*(const v8bf*)p, *(const v8bf*)(p + 8));
}
// two 8-element bf16 groups from LDS (A-fragment; called with p1 = p0 + 8, so k = 16h + s)
static __device__ __forceinline__ v16bf lds8x2(const __bf16* p0, const __bf16* p1) {
  return cat8(*(const v8bf*)p0, *(const v8bf*)(p1));
}
// 16 contiguous f32 from global, converted to bf16 (W B-fragment)
static __device__ __forceinline__ v16bf cvt16(const float* p) {
  v16bf r;
#pragma unroll
  for (int i = 0; i < 16; ++i) r[i] = (__bf16)p[i];
  return r;
}
// two 8-f32 groups converted to bf16 (Q A-fragment; called with p1 = p0 + 8, so k = 16h + s)
static __device__ __forceinline__ v16bf cvt8x2(const float* p0, const float* p1) {
  v16bf r;
#pragma unroll
  for (int i = 0; i < 8; ++i) r[i] = (__bf16)p0[i];
#pragma unroll
  for (int i = 0; i < 8; ++i) r[8 + i] = (__bf16)p1[i];
  return r;
}
// reductions across the 16-lane half-groups of the C layout (wave32)
static __device__ __forceinline__ float redmax16(float v) {
#pragma unroll
  for (int m = 8; m >= 1; m >>= 1) v = fmaxf(v, __shfl_xor(v, m, 32));
  return v;
}
static __device__ __forceinline__ float redsum16(float v) {
#pragma unroll
  for (int m = 8; m >= 1; m >>= 1) v += __shfl_xor(v, m, 32);
  return v;
}

__global__ __launch_bounds__(256)
void sdgcn_fused_kernel(const float* __restrict__ X, const float* __restrict__ A,
                        const float* __restrict__ W, const float* __restrict__ gamma,
                        const float* __restrict__ beta, float* __restrict__ out) {
  __shared__ __bf16 sXR[32 * XR_P];        // X chunk, row-major [m_local][d]
  __shared__ __bf16 sXT[DD * XT_P];        // X chunk, d-major   [d][m_local]
  __shared__ __bf16 sPS[8 * 16 * PS_P];    // per-wave repack scratch [row][k]

  const int tid  = threadIdx.x;
  const int lane = tid & 31;
  const int w    = tid >> 5;        // wave id 0..7
  const int col  = lane & 15;       // C-layout column / A-layout row
  const int hi   = lane >> 4;       // half-group select

  const int bt = blockIdx.x >> 3;           // (b*T + t)
  const int rb = (blockIdx.x & 7) * 128;    // row-tile base in N

  const float inv_a = 0.08838834764831845f; // 1/sqrt(D + 1e-8)
  const float inv_b = 0.08838834764831845f; // 1/sqrt(D + 1e-9)

  const float* Xbt = X + (size_t)bt * NN * DD;

  // ---- Q strip: 16 rows per wave as 4 bf16 A-fragments (K = 4 x 32) ----
  const int qrow = rb + w * 16 + col;
  const float* qp = Xbt + (size_t)qrow * DD;
  v16bf qa[4];
#pragma unroll
  for (int c = 0; c < 4; ++c)
    qa[c] = cvt8x2(qp + 32 * c + 16 * hi, qp + 32 * c + 16 * hi + 8);

  // ---- flash-state ----
  v8f acc[8];
#pragma unroll
  for (int j = 0; j < 8; ++j)
#pragma unroll
    for (int r = 0; r < 8; ++r) acc[j][r] = 0.0f;
  float rmax[8], rsum[8];
#pragma unroll
  for (int r = 0; r < 8; ++r) { rmax[r] = -INFINITY; rsum[r] = 0.0f; }

  __bf16* myps = sPS + w * 16 * PS_P;

  // =================== main loop over 32-column chunks of N ===================
  for (int ch = 0; ch < 32; ++ch) {
    const int m0 = ch * 32;
    __syncthreads();   // previous chunk fully consumed
    {
      // stage X[m0:m0+32, 0:128] -> bf16, both layouts. 256 thr x 16 elems.
      const int ml = tid >> 3;            // 0..31
      const int d0 = (tid & 7) << 4;      // 0,16,...,112
      const float* src = Xbt + (size_t)(m0 + ml) * DD + d0;
      const float4* s4 = (const float4*)src;
      float4 q0 = s4[0], q1 = s4[1], q2 = s4[2], q3 = s4[3];
      float f[16] = {q0.x,q0.y,q0.z,q0.w, q1.x,q1.y,q1.z,q1.w,
                     q2.x,q2.y,q2.z,q2.w, q3.x,q3.y,q3.z,q3.w};
      __bf16 b16[16];
      v8bf lo, hi8;
#pragma unroll
      for (int i = 0; i < 16; ++i) b16[i] = (__bf16)f[i];
#pragma unroll
      for (int i = 0; i < 8; ++i) { lo[i] = b16[i]; hi8[i] = b16[8 + i]; }
      *(v8bf*)(sXR + ml * XR_P + d0)     = lo;
      *(v8bf*)(sXR + ml * XR_P + d0 + 8) = hi8;
#pragma unroll
      for (int e = 0; e < 16; ++e) sXT[(d0 + e) * XT_P + ml] = b16[e];
      if (ch < 31) __builtin_prefetch(src + 32 * DD, 0, 1);  // global_prefetch_b8
    }
    __syncthreads();

    // ---- S = Q . K^T : two 16x16 tiles, K over 4x32 ----
    v8f s0 = {0,0,0,0,0,0,0,0}, s1 = {0,0,0,0,0,0,0,0};
#pragma unroll
    for (int c = 0; c < 4; ++c) {
      v16bf k0 = lds16(sXR + col * XR_P        + 32 * c + 16 * hi);
      v16bf k1 = lds16(sXR + (col + 16) * XR_P + 32 * c + 16 * hi);
      s0 = WMMA_BF16(qa[c], k0, s0);
      s1 = WMMA_BF16(qa[c], k1, s1);
    }

    // ---- adjacency mask values A[row, m] (L2-resident) ----
    float a0[8], a1[8];
#pragma unroll
    for (int r = 0; r < 8; ++r) {
      const int grow = rb + w * 16 + r + 8 * hi;
      const float* ap = A + (size_t)grow * NN + m0 + col;
      a0[r] = ap[0];
      a1[r] = ap[16];
    }

    // ---- online softmax (max/sum over full row, mask applied to numerator) ----
#pragma unroll
    for (int r = 0; r < 8; ++r) {
      float mcur = redmax16(fmaxf(s0[r], s1[r]));
      float mnew = fmaxf(rmax[r], mcur);
      float corr = __expf(rmax[r] - mnew);
      float e0 = __expf(s0[r] - mnew);
      float e1 = __expf(s1[r] - mnew);
      rsum[r] = rsum[r] * corr + redsum16(e0 + e1);
      rmax[r] = mnew;
#pragma unroll
      for (int j = 0; j < 8; ++j) acc[j][r] *= corr;
      s0[r] = e0 * a0[r];   // masked numerator P
      s1[r] = e1 * a1[r];
    }

    // ---- repack P: C-layout -> A-layout through per-wave LDS scratch ----
#pragma unroll
    for (int r = 0; r < 8; ++r) {
      myps[(r + 8 * hi) * PS_P + col]      = (__bf16)s0[r];
      myps[(r + 8 * hi) * PS_P + 16 + col] = (__bf16)s1[r];
    }
    v16bf pf = lds8x2(myps + col * PS_P + 16 * hi,
                      myps + col * PS_P + 16 * hi + 8);

    // ---- acc += P (16x32) . Xchunk (32x128) ----
#pragma unroll
    for (int j = 0; j < 8; ++j) {
      v16bf xv = lds16(sXT + (16 * j + col) * XT_P + 16 * hi);
      acc[j] = WMMA_BF16(pf, xv, acc[j]);
    }
  }

  // ---- finalize h = acc * inv_a / Z, spill strip to bf16 scratch ----
#pragma unroll
  for (int r = 0; r < 8; ++r) {
    const float sc = inv_a / rsum[r];
#pragma unroll
    for (int j = 0; j < 8; ++j)
      myps[(r + 8 * hi) * PS_P + 16 * j + col] = (__bf16)(acc[j][r] * sc);
  }

  // ---- y = h (16x128) . W^T (128x128), W B-fragments straight from L2 ----
  v8f y[8];
#pragma unroll
  for (int j = 0; j < 8; ++j)
#pragma unroll
    for (int r = 0; r < 8; ++r) y[j][r] = 0.0f;
#pragma unroll
  for (int c = 0; c < 4; ++c) {
    v16bf ha = lds8x2(myps + col * PS_P + 32 * c + 16 * hi,
                      myps + col * PS_P + 32 * c + 16 * hi + 8);
#pragma unroll
    for (int j = 0; j < 8; ++j) {
      v16bf wf = cvt16(W + (size_t)(16 * j + col) * DD + 32 * c + 16 * hi);
      y[j] = WMMA_BF16(ha, wf, y[j]);
    }
  }

  // ---- relu * inv_b ----
#pragma unroll
  for (int j = 0; j < 8; ++j)
#pragma unroll
    for (int r = 0; r < 8; ++r) y[j][r] = fmaxf(y[j][r], 0.0f) * inv_b;

  // ---- gamma/beta per output column ----
  float gj[8], bj[8];
#pragma unroll
  for (int j = 0; j < 8; ++j) { gj[j] = gamma[16 * j + col]; bj[j] = beta[16 * j + col]; }

  // ---- LayerNorm over D + residual + store ----
#pragma unroll
  for (int r = 0; r < 8; ++r) {
    float s = 0.0f, sq = 0.0f;
#pragma unroll
    for (int j = 0; j < 8; ++j) { s += y[j][r]; sq += y[j][r] * y[j][r]; }
    s  = redsum16(s);
    sq = redsum16(sq);
    const float mu   = s  * (1.0f / 128.0f);
    const float var  = sq * (1.0f / 128.0f) - mu * mu;
    const float rstd = rsqrtf(var + 1e-5f);
    const int grow = rb + w * 16 + r + 8 * hi;
    const float* xres = Xbt + (size_t)grow * DD;
    float* op = out + (size_t)bt * NN * DD + (size_t)grow * DD;
#pragma unroll
    for (int j = 0; j < 8; ++j) {
      const int o = 16 * j + col;
      op[o] = (y[j][r] - mu) * rstd * gj[j] + bj[j] + xres[o];
    }
  }
}

extern "C" void kernel_launch(void* const* d_in, const int* in_sizes, int n_in,
                              void* d_out, int out_size, void* d_ws, size_t ws_size,
                              hipStream_t stream) {
  (void)in_sizes; (void)n_in; (void)out_size; (void)d_ws; (void)ws_size;
  const float* X     = (const float*)d_in[0];
  const float* A     = (const float*)d_in[1];
  const float* W     = (const float*)d_in[2];
  const float* gamma = (const float*)d_in[3];
  const float* beta  = (const float*)d_in[4];
  float* out = (float*)d_out;

  dim3 grid(BT * (NN / 128));   // 64 (b,t) x 8 row-tiles = 512 workgroups
  dim3 block(256);              // 8 wave32 waves, one 16-row strip each
  sdgcn_fused_kernel<<<grid, block, 0, stream>>>(X, A, W, gamma, beta, out);
}
